// LlamaAttention_82729660056187
// MI455X (gfx1250) — matrix operand order repaired and checked by reading, not compiled, not run on hardware
//
#include <hip/hip_runtime.h>
#include <hip/hip_bf16.h>

// ---------------- Problem constants ----------------
#define BB     2
#define SS     2048
#define HH     4096
#define NHEAD  32
#define NKV    8
#define HDIM   128
#define SCALING 0.08838834764831845f   // 128^-0.5

typedef __attribute__((ext_vector_type(16))) __bf16 v16bf;
typedef __attribute__((ext_vector_type(8)))  float  v8f;

// ---------------- WMMA fragment helpers ----------------
// A-fragment (16x32 bf16, from fp32 row-major source; row = this lane's row)
// lane<16  : K = kk+0..15
// lane>=16 : K = kk+16..31
__device__ __forceinline__ v16bf load_a_frag(const float* __restrict__ rowptr, int kk) {
    const int lane = threadIdx.x & 31;
    const int half = lane >> 4;
    const float4* p0 = reinterpret_cast<const float4*>(rowptr + kk + half * 16);
    const float4* p1 = reinterpret_cast<const float4*>(rowptr + kk + half * 16 + 8);
    float4 a0 = p0[0], a1 = p0[1], b0 = p1[0], b1 = p1[1];
    v16bf a;
    a[0]=(__bf16)a0.x; a[1]=(__bf16)a0.y; a[2]=(__bf16)a0.z; a[3]=(__bf16)a0.w;
    a[4]=(__bf16)a1.x; a[5]=(__bf16)a1.y; a[6]=(__bf16)a1.z; a[7]=(__bf16)a1.w;
    a[8]=(__bf16)b0.x; a[9]=(__bf16)b0.y; a[10]=(__bf16)b0.z; a[11]=(__bf16)b0.w;
    a[12]=(__bf16)b1.x; a[13]=(__bf16)b1.y; a[14]=(__bf16)b1.z; a[15]=(__bf16)b1.w;
    return a;
}

// B-fragment (32x16 bf16) where column n is stored contiguously (W row / K row / Vt row)
// colptr = base of this lane's column; lane<16 holds K=kk+0..15, lane>=16 K=kk+16..31
__device__ __forceinline__ v16bf load_b_frag(const float* __restrict__ colptr, int kk) {
    const int lane = threadIdx.x & 31;
    const int half = lane >> 4;
    const float4* p = reinterpret_cast<const float4*>(colptr + kk + half * 16);
    float4 x0 = p[0], x1 = p[1], x2 = p[2], x3 = p[3];
    v16bf b;
    b[0]=(__bf16)x0.x; b[1]=(__bf16)x0.y; b[2]=(__bf16)x0.z; b[3]=(__bf16)x0.w;
    b[4]=(__bf16)x1.x; b[5]=(__bf16)x1.y; b[6]=(__bf16)x1.z; b[7]=(__bf16)x1.w;
    b[8]=(__bf16)x2.x; b[9]=(__bf16)x2.y; b[10]=(__bf16)x2.z; b[11]=(__bf16)x2.w;
    b[12]=(__bf16)x3.x; b[13]=(__bf16)x3.y; b[14]=(__bf16)x3.z; b[15]=(__bf16)x3.w;
    return b;
}

// B-fragment from LDS: column n stored at row (jtile*16+n16), stride rowStride bf16,
// K offset koff within the row.
__device__ __forceinline__ v16bf load_b_lds(const __bf16* __restrict__ lb, int rowStride,
                                            int jtile, int koff) {
    const int lane = threadIdx.x & 31;
    const int half = lane >> 4;
    const int n16  = lane & 15;
    const __bf16* p = lb + (jtile * 16 + n16) * rowStride + koff + half * 16;
    v16bf b;
#pragma unroll
    for (int i = 0; i < 16; ++i) b[i] = p[i];
    return b;
}

// convert 16 contiguous f32 -> 16 bf16 in LDS
__device__ __forceinline__ void stage16(__bf16* __restrict__ dst,
                                        const float* __restrict__ src) {
    const float4* wp = reinterpret_cast<const float4*>(src);
    float4 w0 = wp[0], w1 = wp[1], w2 = wp[2], w3 = wp[3];
    dst[0]=(__bf16)w0.x; dst[1]=(__bf16)w0.y; dst[2]=(__bf16)w0.z; dst[3]=(__bf16)w0.w;
    dst[4]=(__bf16)w1.x; dst[5]=(__bf16)w1.y; dst[6]=(__bf16)w1.z; dst[7]=(__bf16)w1.w;
    dst[8]=(__bf16)w2.x; dst[9]=(__bf16)w2.y; dst[10]=(__bf16)w2.z; dst[11]=(__bf16)w2.w;
    dst[12]=(__bf16)w3.x; dst[13]=(__bf16)w3.y; dst[14]=(__bf16)w3.z; dst[15]=(__bf16)w3.w;
}

#define LDSB_STRIDE 40   // 32 bf16 + 8 pad -> 20-dword row stride, conflict-free

// ================= Kernel 1: generic X @ W^T GEMM (bf16 WMMA, fp32 acc) =================
// A: [M x Kd] fp32 row-major (lda = Kd), W: [N x Kd] fp32 row-major, C: [M x N] (ldc = N)
// Block = 256 threads (8 waves) computes a 128x128 C tile.
// B tile double-buffered in LDS: next slice's global loads are issued before the WMMA
// chain, cvt+ds_store after it -> the load-wait hides behind 8 WMMAs.
__global__ void __launch_bounds__(256)
gemm_xwT_kernel(const float* __restrict__ A, const float* __restrict__ W,
                float* __restrict__ C, int Kd, int ldc) {
    __shared__ __align__(16) __bf16 lb[2][128 * LDSB_STRIDE];

    const int mt   = blockIdx.x;
    const int nt   = blockIdx.y;
    const int wave = threadIdx.x >> 5;
    const int lane = threadIdx.x & 31;

    const int   arowIdx = mt * 128 + wave * 16 + (lane & 15);
    const float* arow   = A + (size_t)arowIdx * Kd;

    // staging assignment: thread t handles column (nt*128 + t/2), K-halves seg*16
    const int   sn  = threadIdx.x >> 1;
    const int   seg = threadIdx.x & 1;
    const float* wsrc = W + (size_t)(nt * 128 + sn) * Kd + seg * 16;
    const int   soff  = sn * LDSB_STRIDE + seg * 16;

    // prologue: stage K-slice 0 into buffer 0
    stage16(&lb[0][soff], wsrc);

    v8f acc[8];
#pragma unroll
    for (int j = 0; j < 8; ++j) acc[j] = (v8f)(0.0f);

    const int nIter = Kd >> 5;
    for (int it = 0; it < nIter; ++it) {
        const int kk = it << 5;
        __syncthreads();   // staged buf[it&1] visible; prior reads of buf[(it+1)&1] done

        // issue next slice's global loads early
        float4 w0, w1, w2, w3;
        const bool haveNext = (it + 1 < nIter);
        if (haveNext) {
            const float4* wp = reinterpret_cast<const float4*>(wsrc + kk + 32);
            w0 = wp[0]; w1 = wp[1]; w2 = wp[2]; w3 = wp[3];
        }

        const __bf16* cur = lb[it & 1];
        v16bf a = load_a_frag(arow, kk);
        v16bf bfr[8];
#pragma unroll
        for (int j = 0; j < 8; ++j) bfr[j] = load_b_lds(cur, LDSB_STRIDE, j, 0);
#pragma unroll
        for (int j = 0; j < 8; ++j) {
            acc[j] = __builtin_amdgcn_wmma_f32_16x16x32_bf16(
                false, a, false, bfr[j], (short)0, acc[j], false, false);
        }

        // convert + store next slice behind the WMMA chain
        if (haveNext) {
            __bf16* dst = &lb[(it + 1) & 1][soff];
            dst[0]=(__bf16)w0.x; dst[1]=(__bf16)w0.y; dst[2]=(__bf16)w0.z; dst[3]=(__bf16)w0.w;
            dst[4]=(__bf16)w1.x; dst[5]=(__bf16)w1.y; dst[6]=(__bf16)w1.z; dst[7]=(__bf16)w1.w;
            dst[8]=(__bf16)w2.x; dst[9]=(__bf16)w2.y; dst[10]=(__bf16)w2.z; dst[11]=(__bf16)w2.w;
            dst[12]=(__bf16)w3.x; dst[13]=(__bf16)w3.y; dst[14]=(__bf16)w3.z; dst[15]=(__bf16)w3.w;
        }
    }

    const int half = lane >> 4;
    const int n16  = lane & 15;
#pragma unroll
    for (int j = 0; j < 8; ++j) {
#pragma unroll
        for (int r = 0; r < 8; ++r) {
            int row = mt * 128 + wave * 16 + r + 8 * half;
            int col = nt * 128 + j * 16 + n16;
            C[(size_t)row * ldc + col] = acc[j][r];
        }
    }
}

// ================= Kernel 2: RoPE (in place on Q or K projection output) =================
// qk: [B*S][n_heads*128] fp32; cos/sin: [B*S][128]
__global__ void __launch_bounds__(256)
rope_kernel(float* __restrict__ qk, const float* __restrict__ cosp,
            const float* __restrict__ sinp, int n_heads) {
    const int idx = blockIdx.x * 256 + threadIdx.x;   // (bs, h, d<64)
    const int d   = idx & 63;
    const int h   = (idx >> 6) % n_heads;
    const int bs  = idx / (64 * n_heads);
    float* row = qk + (size_t)bs * (n_heads * HDIM) + h * HDIM;
    const float* c  = cosp + (size_t)bs * HDIM;
    const float* sn = sinp + (size_t)bs * HDIM;
    float x1 = row[d], x2 = row[d + 64];
    row[d]      = x1 * c[d]      - x2 * sn[d];
    row[d + 64] = x2 * c[d + 64] + x1 * sn[d + 64];
}

// ================= Kernel 3: transpose V -> Vt[b][kv][d][s] =================
__global__ void __launch_bounds__(256)
transpose_v_kernel(const float* __restrict__ V, float* __restrict__ Vt) {
    const int idx = blockIdx.x * 256 + threadIdx.x;   // total = B*NKV*HDIM*SS
    const int s   = idx & (SS - 1);
    const int d   = (idx >> 11) & (HDIM - 1);
    const int kvb = idx >> 18;              // b*NKV + kv
    const int b   = kvb >> 3;
    const int kv  = kvb & 7;
    Vt[idx] = V[((size_t)(b * SS + s)) * (NKV * HDIM) + kv * HDIM + d];
}

// ================= Kernel 4: scores = scale * Q @ K^T (causal tiles only) =================
// Q: [B*S][NHEAD*128], K: [B*S][NKV*128], Wts: [B][NHEAD][S][S]
// The 128x128 K-tile is shared by all 8 waves -> stage it once in LDS as bf16
// (converted once instead of 8x, ~34KB of the 320KB/WGP pool).
#define KTS 136    // 128 bf16 + 8 pad -> 68-dword row stride, conflict-free
__global__ void __launch_bounds__(256)
attn_scores_kernel(const float* __restrict__ Q, const float* __restrict__ K,
                   float* __restrict__ Wts) {
    const int qt = blockIdx.x;
    const int kt = blockIdx.y;
    if (kt > qt) return;                     // uniform per block; above diagonal never read
    const int bh = blockIdx.z;
    const int b  = bh >> 5;
    const int h  = bh & 31;
    const int kv = h >> 2;

    __shared__ __align__(16) __bf16 klds[128 * KTS];

    // stage K tile: thread t handles kpos = t/2, d-range seg*64..+63
    {
        const int sn  = threadIdx.x >> 1;
        const int seg = threadIdx.x & 1;
        const float* ksrc =
            K + (size_t)(b * SS + kt * 128 + sn) * (NKV * HDIM) + kv * HDIM + seg * 64;
        __bf16* dst = &klds[sn * KTS + seg * 64];
#pragma unroll
        for (int c = 0; c < 4; ++c) stage16(dst + c * 16, ksrc + c * 16);
    }
    __syncthreads();

    const int wave = threadIdx.x >> 5;
    const int lane = threadIdx.x & 31;
    const int n16  = lane & 15;

    const int    qrow = qt * 128 + wave * 16 + n16;
    const float* arow = Q + (size_t)(b * SS + qrow) * (NHEAD * HDIM) + h * HDIM;

    v8f acc[8];
#pragma unroll
    for (int j = 0; j < 8; ++j) acc[j] = (v8f)(0.0f);

#pragma unroll
    for (int kk = 0; kk < HDIM; kk += 32) {
        v16bf a = load_a_frag(arow, kk);
        v16bf bfr[8];
#pragma unroll
        for (int j = 0; j < 8; ++j) bfr[j] = load_b_lds(klds, KTS, j, kk);
#pragma unroll
        for (int j = 0; j < 8; ++j) {
            acc[j] = __builtin_amdgcn_wmma_f32_16x16x32_bf16(
                false, a, false, bfr[j], (short)0, acc[j], false, false);
        }
    }

    const int half = lane >> 4;
    float* cbase = Wts + ((size_t)bh * SS + qt * 128 + wave * 16) * SS + kt * 128;
#pragma unroll
    for (int j = 0; j < 8; ++j) {
#pragma unroll
        for (int r = 0; r < 8; ++r) {
            cbase[(size_t)(r + 8 * half) * SS + j * 16 + n16] = acc[j][r] * SCALING;
        }
    }
}

// ================= Kernel 5: causal row softmax (in place on Wts) =================
__global__ void __launch_bounds__(256)
softmax_rows_kernel(float* __restrict__ Wts) {
    const int row = blockIdx.x;              // (b*NHEAD + h)*S + q
    const int q   = row & (SS - 1);
    float* p = Wts + (size_t)row * SS;
    const int t = threadIdx.x;
    __shared__ float red[256];

    float mx = -3.0e38f;
    for (int j = t; j <= q; j += 256) mx = fmaxf(mx, p[j]);
    red[t] = mx; __syncthreads();
    for (int off = 128; off > 0; off >>= 1) {
        if (t < off) red[t] = fmaxf(red[t], red[t + off]);
        __syncthreads();
    }
    mx = red[0]; __syncthreads();

    float sum = 0.0f;
    for (int j = t; j <= q; j += 256) {
        float e = __expf(p[j] - mx);
        p[j] = e;
        sum += e;
    }
    red[t] = sum; __syncthreads();
    for (int off = 128; off > 0; off >>= 1) {
        if (t < off) red[t] += red[t + off];
        __syncthreads();
    }
    const float inv = 1.0f / red[0];

    for (int j = t; j <= q; j += 256) p[j] *= inv;
    for (int j = q + 1 + t; j < SS; j += 256) p[j] = 0.0f;   // zero-fill masked region
}

// ================= Kernel 6: attn_out = P @ V (causal-clipped K loop) =================
// Wts: [B][NHEAD][S][S], Vt: [B][NKV][128][S], AO: [B*S][NHEAD*128]
// Vt (16.7MB) is L2-resident; preload all 8 B fragments before the WMMA chain.
__global__ void __launch_bounds__(256)
attn_pv_kernel(const float* __restrict__ Wts, const float* __restrict__ Vt,
               float* __restrict__ AO) {
    const int qt = blockIdx.x;
    const int bh = blockIdx.y;
    const int b  = bh >> 5;
    const int h  = bh & 31;
    const int kv = h >> 2;

    const int wave = threadIdx.x >> 5;
    const int lane = threadIdx.x & 31;
    const int n16  = lane & 15;

    const int    qrow = qt * 128 + wave * 16 + n16;
    const float* arow = Wts + ((size_t)bh * SS + qrow) * SS;
    const float* vcol = Vt + ((size_t)(b * NKV + kv) * HDIM + n16) * SS;  // + j*16*SS

    v8f acc[8];
#pragma unroll
    for (int j = 0; j < 8; ++j) acc[j] = (v8f)(0.0f);

    const int kmax = (qt + 1) * 128;         // P is zero beyond the diagonal tile row
    for (int kk = 0; kk < kmax; kk += 32) {
        v16bf a = load_a_frag(arow, kk);
        v16bf bfr[8];
#pragma unroll
        for (int j = 0; j < 8; ++j) bfr[j] = load_b_frag(vcol + (size_t)j * 16 * SS, kk);
#pragma unroll
        for (int j = 0; j < 8; ++j) {
            acc[j] = __builtin_amdgcn_wmma_f32_16x16x32_bf16(
                false, a, false, bfr[j], (short)0, acc[j], false, false);
        }
    }

    const int half = lane >> 4;
#pragma unroll
    for (int j = 0; j < 8; ++j) {
#pragma unroll
        for (int r = 0; r < 8; ++r) {
            int row = qt * 128 + wave * 16 + r + 8 * half;
            AO[(size_t)(b * SS + row) * HH + h * HDIM + j * 16 + n16] = acc[j][r];
        }
    }
}

// ================= host-side launch =================
extern "C" void kernel_launch(void* const* d_in, const int* in_sizes, int n_in,
                              void* d_out, int out_size, void* d_ws, size_t ws_size,
                              hipStream_t stream) {
    const float* X    = (const float*)d_in[0];   // hidden_states [B][S][H]
    const float* cosp = (const float*)d_in[1];   // [B][S][128]
    const float* sinp = (const float*)d_in[2];   // [B][S][128]
    // d_in[3] attention_mask: pure causal, applied analytically
    const float* Wq = (const float*)d_in[4];     // [4096][4096]
    const float* Wk = (const float*)d_in[5];     // [1024][4096]
    const float* Wv = (const float*)d_in[6];     // [1024][4096]
    const float* Wo = (const float*)d_in[7];     // [4096][4096]

    float* out = (float*)d_out;                          // attn_output: B*S*H
    float* wts = out + (size_t)BB * SS * HH;             // attn_weights: B*32*S*S

    float* ws = (float*)d_ws;
    float* Qb = ws;                          // B*S*NHEAD*128 = 16,777,216 f32
    float* Kb = Qb + (size_t)16777216;       // B*S*NKV*128   =  4,194,304
    float* Vb = Kb + (size_t)4194304;        //                  4,194,304
    float* Vt = Vb + (size_t)4194304;        //                  4,194,304
    float* AO = Vt + (size_t)4194304;        //                 16,777,216

    dim3 blk(256);

    // QKV projections (X @ W^T)
    gemm_xwT_kernel<<<dim3(32, 32), blk, 0, stream>>>(X, Wq, Qb, HH, NHEAD * HDIM);
    gemm_xwT_kernel<<<dim3(32, 8),  blk, 0, stream>>>(X, Wk, Kb, HH, NKV * HDIM);
    gemm_xwT_kernel<<<dim3(32, 8),  blk, 0, stream>>>(X, Wv, Vb, HH, NKV * HDIM);

    // RoPE on Q and K
    rope_kernel<<<32768, blk, 0, stream>>>(Qb, cosp, sinp, NHEAD);
    rope_kernel<<<8192,  blk, 0, stream>>>(Kb, cosp, sinp, NKV);

    // V -> Vt[b][kv][d][s]
    transpose_v_kernel<<<16384, blk, 0, stream>>>(Vb, Vt);

    // scores (lower-triangular tiles), softmax, P@V
    attn_scores_kernel<<<dim3(16, 16, 64), blk, 0, stream>>>(Qb, Kb, wts);
    softmax_rows_kernel<<<BB * NHEAD * SS, blk, 0, stream>>>(wts);
    attn_pv_kernel<<<dim3(16, 64), blk, 0, stream>>>(wts, Vt, AO);

    // output projection
    gemm_xwT_kernel<<<dim3(32, 32), blk, 0, stream>>>(AO, Wo, out, HH, HH);
}
